// ScaledDotProductAttention_6399501271336
// MI455X (gfx1250) — matrix operand order repaired and checked by reading, not compiled, not run on hardware
//
#include <hip/hip_runtime.h>

typedef __attribute__((ext_vector_type(16))) _Float16 v16h;
typedef __attribute__((ext_vector_type(8)))  float    v8f;
typedef __attribute__((ext_vector_type(4)))  float    f4;

#define B_   4
#define H_   16
#define S_   2048
#define D_   64
#define QT   16            // query rows per workgroup tile
#define ROWF (S_ + 4)      // padded LDS row stride in floats (bank-conflict pad, 16B-aligned rows)
#define NEG_INF (-__builtin_huge_valf())

// D = A*B + C  with A = 16x32 f16, B = 32x16 f16, C/D = 16x16 f32.
// Per CDNA5 ISA 7.12.2 (wave32):
//   A: lane holds row M = lane&15; lanes 0-15 cover K base 0, lanes 16-31 K base 8;
//      element j -> K = kbaseA + (j<8 ? j : j+8)        (kbaseA = (lane>>4)*8)
//   B (here, and A too, use contiguous k): col N = lane&15; lanes 0-15 K=0..15, lanes 16-31 K=16..31;
//      element j -> K = kbaseB + j                      (kbaseB = (lane>>4)*16)
//   C/D: element r -> M = r + (lane>>4)*8, N = lane&15

__global__ __launch_bounds__(128)
void attn_sdpa_kernel(const float* __restrict__ Q, const float* __restrict__ K,
                      const float* __restrict__ V, const int* __restrict__ mask,
                      float* __restrict__ out, float* __restrict__ attnW)
{
    extern __shared__ float smem[];              // [QT * ROWF] score tile
    float* red   = smem + QT * ROWF;             // [QT*8] reduction scratch
    float* rstat = red  + QT * 8;                // [QT]   1/rowsum

    const int tid  = threadIdx.x;
    const int lane = tid & 31;
    const int wave = tid >> 5;                   // 0..3

    const int qt = blockIdx.x & 127;             // q-tile within (b,h)
    const int bh = blockIdx.x >> 7;              // b*H + h
    const int b  = bh >> 4;
    const int qbase = qt * QT;

    const size_t qkvBase  = (size_t)bh * S_ * D_;
    const size_t maskBase = (size_t)b  * S_ * S_;
    const float  scale    = 0.125f;              // 1/sqrt(64)

    const int ln15   = lane & 15;
    const int hi     = lane >> 4;                // 0 or 1
    const int kbaseA = hi * 8;
    const int kbaseB = hi * 16;

    // ---- Q fragments: d in [0,32) and [32,64) ------------------------------
    v16h a0, a1;
    {
        const float* qrow = Q + qkvBase + (size_t)(qbase + ln15) * D_;
        #pragma unroll
        for (int j = 0; j < 16; ++j) {
            const int k = kbaseB + j;
            a0[j] = (_Float16)qrow[k];
            a1[j] = (_Float16)qrow[k + 32];
        }
    }

    // ---- Phase 1: scores = scale * Q K^T, masked, into LDS -----------------
    for (int t = wave; t < S_ / 16; t += 4) {
        const float* krow = K + qkvBase + (size_t)(t * 16 + ln15) * D_;
        if (t + 4 < S_ / 16)                      // hint the next K tile stream
            __builtin_prefetch(krow + 4 * 16 * D_, 0, 0);
        v16h b0, b1;
        #pragma unroll
        for (int j = 0; j < 16; ++j) {
            b0[j] = (_Float16)krow[kbaseB + j];
            b1[j] = (_Float16)krow[kbaseB + j + 32];
        }
        v8f c = {};
        c = __builtin_amdgcn_wmma_f32_16x16x32_f16(false, a0, false, b0, (short)0, c, false, false);
        c = __builtin_amdgcn_wmma_f32_16x16x32_f16(false, a1, false, b1, (short)0, c, false, false);
        const int n = t * 16 + ln15;
        #pragma unroll
        for (int r = 0; r < 8; ++r) {
            const int m  = r + hi * 8;
            const int mv = mask[maskBase + (size_t)(qbase + m) * S_ + n];
            smem[m * ROWF + n] = (mv == 0) ? NEG_INF : c[r] * scale;
        }
    }
    __syncthreads();

    // ---- Phase 2: row softmax over LDS-resident 16x2048 tile ---------------
    // 8 threads/row, each owns 256 floats = 64 float4 (16B-aligned chunks).
    {
        const int row = tid >> 3, sub = tid & 7;
        f4* prow4 = (f4*)(smem + row * ROWF) + sub * 64;

        f4 mxv = { NEG_INF, NEG_INF, NEG_INF, NEG_INF };
        #pragma unroll 4
        for (int i = 0; i < 64; ++i) {
            const f4 v = prow4[i];
            mxv.x = fmaxf(mxv.x, v.x); mxv.y = fmaxf(mxv.y, v.y);
            mxv.z = fmaxf(mxv.z, v.z); mxv.w = fmaxf(mxv.w, v.w);
        }
        red[row * 8 + sub] = fmaxf(fmaxf(mxv.x, mxv.y), fmaxf(mxv.z, mxv.w));
        __syncthreads();
        float rmax = red[row * 8 + 0];
        #pragma unroll
        for (int j = 1; j < 8; ++j) rmax = fmaxf(rmax, red[row * 8 + j]);
        __syncthreads();                            // red[] reused for sums

        f4 sv = { 0.f, 0.f, 0.f, 0.f };
        #pragma unroll 4
        for (int i = 0; i < 64; ++i) {
            f4 v = prow4[i];
            v.x = __expf(v.x - rmax); v.y = __expf(v.y - rmax);
            v.z = __expf(v.z - rmax); v.w = __expf(v.w - rmax);
            prow4[i] = v;                           // keep UNnormalized in LDS
            sv += v;
        }
        red[row * 8 + sub] = (sv.x + sv.y) + (sv.z + sv.w);
        __syncthreads();
        if (sub == 0) {
            float tot = 0.f;
            #pragma unroll
            for (int j = 0; j < 8; ++j) tot += red[row * 8 + j];
            rstat[row] = 1.0f / tot;
        }
        __syncthreads();
    }

    // ---- Phase 3a: stream normalized P to attn_weights (float4, coalesced) -
    {
        const size_t gbase = ((size_t)bh * S_ + qbase) * (size_t)S_;
        #pragma unroll 4
        for (int k = 0; k < (QT * S_ / 4) / 128; ++k) {   // 64 iters
            const int i4  = tid + 128 * k;
            const int row = i4 >> 9;                      // 512 float4 per row
            const int c4  = i4 & 511;
            f4 v = *(const f4*)(smem + row * ROWF + c4 * 4);
            v *= rstat[row];
            *(f4*)(attnW + gbase + (size_t)row * S_ + c4 * 4) = v;
        }
    }

    // ---- Phase 3b: O = diag(1/rowsum) * P @ V via WMMA ---------------------
    {
        const int ncol = wave * 16 + ln15;                // each wave: 16 cols of D
        v8f c = {};
        for (int kt = 0; kt < S_ / 32; ++kt) {            // 64 WMMAs per wave
            v16h a, bv;
            const float* pr = smem + ln15 * ROWF + kt * 32;
            #pragma unroll
            for (int j = 0; j < 16; ++j) {
                const int kk = kbaseB + j;
                a[j] = (_Float16)pr[kk];
            }
            const float* vcol = V + qkvBase + (size_t)(kt * 32 + kbaseB) * D_ + ncol;
            #pragma unroll
            for (int j = 0; j < 16; ++j)
                bv[j] = (_Float16)vcol[(size_t)j * D_];
            c = __builtin_amdgcn_wmma_f32_16x16x32_f16(false, a, false, bv, (short)0, c, false, false);
        }
        const size_t obase = ((size_t)bh * S_ + qbase) * (size_t)D_;
        #pragma unroll
        for (int r = 0; r < 8; ++r) {
            const int m = r + hi * 8;
            out[obase + (size_t)m * D_ + ncol] = c[r] * rstat[m];
        }
    }
}

extern "C" void kernel_launch(void* const* d_in, const int* in_sizes, int n_in,
                              void* d_out, int out_size, void* d_ws, size_t ws_size,
                              hipStream_t stream) {
    const float* Q    = (const float*)d_in[0];
    const float* K    = (const float*)d_in[1];
    const float* V    = (const float*)d_in[2];
    const int*   mask = (const int*)d_in[3];

    float* out   = (float*)d_out;                          // [B,H,S,D] first
    float* attnW = out + (size_t)B_ * H_ * S_ * D_;        // then [B,H,S,S]

    const size_t ldsBytes = (size_t)(QT * ROWF + QT * 8 + QT) * sizeof(float); // ~129 KB
    dim3 grid(B_ * H_ * (S_ / QT));                         // 8192 workgroups
    attn_sdpa_kernel<<<grid, 128, ldsBytes, stream>>>(Q, K, V, mask, out, attnW);
}
